// ThreeBodyModel_85581518340208
// MI455X (gfx1250) — matrix operand order repaired and checked by reading, not compiled, not run on hardware
//
#include <hip/hip_runtime.h>
#include <hip/hip_bf16.h>

// ---------------------------------------------------------------------------
// Neural-ODE RK4 (B=1024, T=100, D=12), MLP 12->300->100->50->12 leaky-relu.
// All weights live in LDS (f16, transposed, zero-padded); each wave owns a
// 16-row batch tile and drives the full 99-step recurrence with
// v_wmma_f32_16x16x32_f16 tiles. Fragments fed by ds_load_b128 pairs.
// N-tiles processed in blocks of up to 7 independent accumulators so the
// scheduler fills the WMMA->VALU hazard window (4 co-exec slots for f16 WMMA)
// with other WMMAs instead of v_nops. Accumulators start as zero so the
// first WMMA of each chain uses the inline C=0 operand (no register
// materialization); bias + leaky-relu live in the epilogue where they
// co-execute with the next block's XDL WMMAs.
// ---------------------------------------------------------------------------

typedef __attribute__((ext_vector_type(16))) _Float16 v16h;
typedef __attribute__((ext_vector_type(8)))  float    v8f;
typedef __attribute__((ext_vector_type(4)))  int      v4i;

namespace cfg {
constexpr int B = 1024, T = 100, D = 12;
// padded layer dims (K padded to x32 for the WMMA K-loop, N padded to x16)
constexpr int S0 = 40;    // act0 row stride (halves): K1=32 + 8 skew
constexpr int S1 = 328;   // act1 row stride: K2=320 + 8
constexpr int S2 = 136;   // act2 row stride: K3=128 + 8
constexpr int S3 = 72;    // act3 row stride: K4=64 + 8
// transposed weights in LDS: row = output-neuron n, stride = input Kpad+8
constexpr int W1T = 0;              constexpr int W1T_SZ = 304 * S0;  // 12160
constexpr int W2T = W1T + W1T_SZ;   constexpr int W2T_SZ = 112 * S1;  // 36736
constexpr int W3T = W2T + W2T_SZ;   constexpr int W3T_SZ = 64  * S2;  // 8704
constexpr int W4T = W3T + W3T_SZ;   constexpr int W4T_SZ = 16  * S3;  // 1152
constexpr int WEIGHT_HALVES = W4T + W4T_SZ;                           // 58752
constexpr int BIAS_BYTE_OFF = WEIGHT_HALVES * 2;                      // 117504
constexpr int BIASF = BIAS_BYTE_OFF / 4;        // float index of bias region
constexpr int B1F = 0, B2F = 304, B3F = 416, B4F = 480;  // bias sub-offsets
constexpr int BIAS_FLOATS = 496;
constexpr int ACT_BYTE_OFF = BIAS_BYTE_OFF + BIAS_FLOATS * 4;         // 119488
// per-wave activation scratch (halves): act0 16xS0, act1 16xS1, act2, act3
constexpr int A0 = 0;
constexpr int A1 = A0 + 16 * S0;      // 640
constexpr int A2 = A1 + 16 * S1;      // 5888
constexpr int A3 = A2 + 16 * S2;      // 8064
constexpr int ACT_HALVES = A3 + 16 * S3;              // 9216 halves
constexpr int ACT_BYTES  = ACT_HALVES * 2;            // 18432 B / wave
constexpr int WAVES_PER_WG = 2;
constexpr int LDS_BYTES = ACT_BYTE_OFF + WAVES_PER_WG * ACT_BYTES;    // 156352
}  // namespace cfg

union FragU {
  v4i  i[2];
  v16h h;
};

// A fragment (16x32 f16, row-major source): lane holds row m = lane&15,
// 16 consecutive K at kb + ko, ko = lane<16 ? 0 : 16.
__device__ __forceinline__ v16h loadA(const _Float16* act, int strideH, int m,
                                      int kb, int ko) {
  const _Float16* p = act + m * strideH + kb + ko;
  FragU u;
  u.i[0] = *(const v4i*)(p);
  u.i[1] = *(const v4i*)(p + 8);
  return u.h;
}

// B fragment (32x16 f16) from transposed weights: lane holds col n = lane&15,
// 16 consecutive K at kb + (lane<16 ? 0 : 16).
__device__ __forceinline__ v16h loadB(const _Float16* wT, int strideH, int nb,
                                      int kb, int n, int ko) {
  const _Float16* p = wT + (nb + n) * strideH + kb + ko;
  FragU u;
  u.i[0] = *(const v4i*)(p);
  u.i[1] = *(const v4i*)(p + 8);
  return u.h;
}

// One hidden layer: out = leaky_relu(in @ W + b), staged through LDS.
// N-tiles processed in blocks of NB with independent zero-init accumulators.
template <int KC, int NT, int NB, bool LEAKY>
__device__ __forceinline__ void layer_fwd(const _Float16* actIn, int sIn,
                                          const _Float16* wT, int sW,
                                          const float* bias, _Float16* actOut,
                                          int sOut, int lane) {
  const int m = lane & 15;
  const int koA = (lane < 16) ? 0 : 16;
  const int n = lane & 15;
  const int koB = (lane < 16) ? 0 : 16;
  const int rowOff = (lane < 16) ? 0 : 8;

#pragma unroll
  for (int t0 = 0; t0 < NT; t0 += NB) {
    v8f acc[NB];
#pragma unroll
    for (int j = 0; j < NB; ++j) {
#pragma unroll
      for (int r = 0; r < 8; ++r) acc[j][r] = 0.0f;  // folds to inline C=0
    }
#pragma unroll
    for (int kc = 0; kc < KC; ++kc) {
      v16h a = loadA(actIn, sIn, m, kc * 32, koA);
#pragma unroll
      for (int j = 0; j < NB; ++j) {
        if (t0 + j < NT) {
          v16h b = loadB(wT, sW, (t0 + j) * 16, kc * 32, n, koB);
          acc[j] = __builtin_amdgcn_wmma_f32_16x16x32_f16(
              false, a, false, b, (short)0, acc[j], false, false);
        }
      }
    }
#pragma unroll
    for (int j = 0; j < NB; ++j) {
      if (t0 + j < NT) {
        float bv = bias[(t0 + j) * 16 + n];
#pragma unroll
        for (int r = 0; r < 8; ++r) {
          float v = acc[j][r] + bv;
          if (LEAKY) v = (v > 0.0f) ? v : 0.01f * v;  // leaky_relu
          actOut[(r + rowOff) * sOut + (t0 + j) * 16 + n] = (_Float16)v;
        }
      }
    }
  }
}

// Output layer (single N=16 tile, no activation): returns C/D-layout v8f.
template <int KC>
__device__ __forceinline__ v8f layer_out(const _Float16* actIn, int sIn,
                                         const _Float16* wT, int sW,
                                         const float* bias, int lane) {
  const int m = lane & 15;
  const int koA = (lane < 16) ? 0 : 16;
  const int n = lane & 15;
  const int koB = (lane < 16) ? 0 : 16;
  v8f acc;
#pragma unroll
  for (int r = 0; r < 8; ++r) acc[r] = 0.0f;
#pragma unroll
  for (int kc = 0; kc < KC; ++kc) {
    v16h a = loadA(actIn, sIn, m, kc * 32, koA);
    v16h b = loadB(wT, sW, 0, kc * 32, n, koB);
    acc = __builtin_amdgcn_wmma_f32_16x16x32_f16(
        false, a, false, b, (short)0, acc, false, false);
  }
  float bv = bias[n];
#pragma unroll
  for (int r = 0; r < 8; ++r) acc[r] += bv;
  return acc;
}

// Write an RK4 stage input (C/D layout v8f) into act0 (f16).
__device__ __forceinline__ void store_state(_Float16* act0, const v8f& xs,
                                            int lane) {
  const int n = lane & 15;
  const int rowOff = (lane < 16) ? 0 : 8;
#pragma unroll
  for (int r = 0; r < 8; ++r)
    act0[(r + rowOff) * cfg::S0 + n] = (_Float16)xs[r];
}

__device__ __forceinline__ v8f mlp_eval(_Float16* act0, _Float16* act1,
                                        _Float16* act2, _Float16* act3,
                                        const _Float16* sm, const float* bias,
                                        int lane) {
  // NB chosen so >=4 independent WMMAs are in flight per block.
  layer_fwd<1, 19, 4, true>(act0, cfg::S0, sm + cfg::W1T, cfg::S0,
                            bias + cfg::B1F, act1, cfg::S1, lane);
  layer_fwd<10, 7, 7, true>(act1, cfg::S1, sm + cfg::W2T, cfg::S1,
                            bias + cfg::B2F, act2, cfg::S2, lane);
  layer_fwd<4, 4, 4, true>(act2, cfg::S2, sm + cfg::W3T, cfg::S2,
                           bias + cfg::B3F, act3, cfg::S3, lane);
  return layer_out<2>(act3, cfg::S3, sm + cfg::W4T, cfg::S3, bias + cfg::B4F,
                      lane);
}

__global__ __launch_bounds__(cfg::WAVES_PER_WG * 32) void node_rk4_kernel(
    const float* __restrict__ y0, const float* __restrict__ t,
    const float* __restrict__ W1, const float* __restrict__ b1,
    const float* __restrict__ W2, const float* __restrict__ b2,
    const float* __restrict__ W3, const float* __restrict__ b3,
    const float* __restrict__ W4, const float* __restrict__ b4,
    float* __restrict__ out) {
  using namespace cfg;
  extern __shared__ __attribute__((aligned(16))) char smemRaw[];
  _Float16* sm = (_Float16*)smemRaw;
  float* smF = (float*)smemRaw;

  const int tid = threadIdx.x;
  const int lane = tid & 31;
  const int wave = tid >> 5;
  const int nthr = WAVES_PER_WG * 32;

  // ---- zero weight+bias region (covers all K/N zero padding) -------------
  for (int i = tid; i < ACT_BYTE_OFF / 16; i += nthr)
    ((v4i*)smemRaw)[i] = (v4i)0;
  __syncthreads();

  // ---- stage weights transposed (N-major, K contiguous) as f16 -----------
  for (int i = tid; i < 12 * 300; i += nthr) {
    int k = i / 300, n = i % 300;
    sm[W1T + n * S0 + k] = (_Float16)W1[i];
  }
  for (int i = tid; i < 300 * 100; i += nthr) {
    int k = i / 100, n = i % 100;
    sm[W2T + n * S1 + k] = (_Float16)W2[i];
  }
  for (int i = tid; i < 100 * 50; i += nthr) {
    int k = i / 50, n = i % 50;
    sm[W3T + n * S2 + k] = (_Float16)W3[i];
  }
  for (int i = tid; i < 50 * 12; i += nthr) {
    int k = i / 12, n = i % 12;
    sm[W4T + n * S3 + k] = (_Float16)W4[i];
  }
  for (int i = tid; i < 300; i += nthr) smF[BIASF + B1F + i] = b1[i];
  for (int i = tid; i < 100; i += nthr) smF[BIASF + B2F + i] = b2[i];
  for (int i = tid; i < 50; i += nthr) smF[BIASF + B3F + i] = b3[i];
  for (int i = tid; i < 12; i += nthr) smF[BIASF + B4F + i] = b4[i];

  // ---- zero this wave's activation scratch (pad columns stay 0) ----------
  char* actBase = smemRaw + ACT_BYTE_OFF + wave * ACT_BYTES;
  for (int i = lane; i < ACT_BYTES / 16; i += 32) ((v4i*)actBase)[i] = (v4i)0;
  __syncthreads();

  _Float16* act0 = (_Float16*)actBase + A0;
  _Float16* act1 = (_Float16*)actBase + A1;
  _Float16* act2 = (_Float16*)actBase + A2;
  _Float16* act3 = (_Float16*)actBase + A3;
  const float* bias = smF + BIASF;

  // ---- per-wave batch tile: rows mBase .. mBase+15 -----------------------
  const int mBase = blockIdx.x * (WAVES_PER_WG * 16) + wave * 16;
  const int n = lane & 15;
  const int rowOff = (lane < 16) ? 0 : 8;

  // x in WMMA C/D layout: reg r holds (row = r+rowOff, col = n)
  v8f x;
#pragma unroll
  for (int r = 0; r < 8; ++r) x[r] = 0.0f;
#pragma unroll
  for (int r = 0; r < 8; ++r) {
    if (n < D) {
      size_t b = (size_t)(mBase + r + rowOff);
      x[r] = y0[b * (size_t)(T * D) + n];
      out[b * (size_t)(T * D) + n] = x[r];  // trajectory at t=0
    }
  }

#pragma unroll 1
  for (int step = 0; step < T - 1; ++step) {
    const float dt = t[step + 1] - t[step];  // row 0 of t
    v8f ksum, kprev;
#pragma unroll
    for (int r = 0; r < 8; ++r) { ksum[r] = 0.0f; kprev[r] = 0.0f; }
#pragma unroll 1
    for (int s = 0; s < 4; ++s) {
      const float a = (s == 0) ? 0.0f : ((s == 3) ? dt : 0.5f * dt);
      const float w = (s == 1 || s == 2) ? 2.0f : 1.0f;
      v8f xs;
#pragma unroll
      for (int r = 0; r < 8; ++r) xs[r] = x[r] + a * kprev[r];
      store_state(act0, xs, lane);  // same-wave LDS ordering is in-order
      v8f k = mlp_eval(act0, act1, act2, act3, sm, bias, lane);
#pragma unroll
      for (int r = 0; r < 8; ++r) ksum[r] += w * k[r];
      kprev = k;
    }
    const float c = dt * (1.0f / 6.0f);
#pragma unroll
    for (int r = 0; r < 8; ++r) x[r] += c * ksum[r];

    if (n < D) {
#pragma unroll
      for (int r = 0; r < 8; ++r) {
        size_t b = (size_t)(mBase + r + rowOff);
        out[b * (size_t)(T * D) + (size_t)((step + 1) * D) + n] = x[r];
      }
    }
  }
}

extern "C" void kernel_launch(void* const* d_in, const int* in_sizes, int n_in,
                              void* d_out, int out_size, void* d_ws,
                              size_t ws_size, hipStream_t stream) {
  (void)in_sizes; (void)n_in; (void)out_size; (void)d_ws; (void)ws_size;
  const float* y0 = (const float*)d_in[0];
  const float* t  = (const float*)d_in[1];
  const float* W1 = (const float*)d_in[2];
  const float* b1 = (const float*)d_in[3];
  const float* W2 = (const float*)d_in[4];
  const float* b2 = (const float*)d_in[5];
  const float* W3 = (const float*)d_in[6];
  const float* b3 = (const float*)d_in[7];
  const float* W4 = (const float*)d_in[8];
  const float* b4 = (const float*)d_in[9];
  float* out = (float*)d_out;

  (void)hipFuncSetAttribute((const void*)node_rk4_kernel,
                            hipFuncAttributeMaxDynamicSharedMemorySize,
                            cfg::LDS_BYTES);
  dim3 grid(cfg::B / (cfg::WAVES_PER_WG * 16));
  dim3 block(cfg::WAVES_PER_WG * 32);
  node_rk4_kernel<<<grid, block, cfg::LDS_BYTES, stream>>>(
      y0, t, W1, b1, W2, b2, W3, b3, W4, b4, out);
}
